// GaussianVoxelizer_111669150044
// MI455X (gfx1250) — compile-verified, matrix operand order checked by reading, not run on hardware
//
#include <hip/hip_runtime.h>
#include <hip/hip_bf16.h>

typedef __attribute__((ext_vector_type(16))) _Float16 v16h;
typedef __attribute__((ext_vector_type(8)))  float    v8f;

#define GP_STRIDE 12          // floats per gaussian (16B-aligned record)
#define WAVES_PER_BLOCK 8
// 0.5 * log2(e): exp(-0.5*maha) == exp2(-C*maha); folded into the Cholesky factor.
#define MAHA_SCALE 0.7213475204444817f
#define MAHA_CUT   (4.0f * MAHA_SCALE)   // maha<=4 in scaled space
// ds_swizzle group-of-32 pattern SWAPX16: xor_mask=0x10, or_mask=0, and_mask=0x1f
#define SWIZ_SWAPX16 0x401F

// ---------------------------------------------------------------------------
// Kernel 1: per-gaussian precompute.
//  - inverse covariance of R diag(s^2) R^T (analytic 3x3 SPD inverse)
//  - scale by 0.5*log2(e), then upper Cholesky U (M' = U^T U) so the hot loop
//    computes maha' = |U diff|^2 and w = opa * exp2(-maha')
//  - record layout (stride 12 floats, 16B aligned):
//      [0..2]=mean, [3]=opacity, [4..7]=u00,u01,u02,u11, [8..9]=u12,u22
//  - pre-swizzle features into f16 B fragments for V_WMMA_F32_16X16X32_F16
//    (ISA 7.12.2 16-bit B layout), dwords indexed [kb][nt][lane][v] so a lane
//    loads 8 contiguous dwords (32B aligned) per fragment.
// ---------------------------------------------------------------------------
__global__ void gv_precompute(const float* __restrict__ means,   // (G,3)
                              const float* __restrict__ opac,    // (G,1)
                              const float* __restrict__ feats,   // (G,D)
                              const float* __restrict__ scales,  // (G,3)
                              const float* __restrict__ rots,    // (G,4) wxyz
                              float* __restrict__ wsG,           // (G,GP_STRIDE)
                              unsigned int* __restrict__ bfrag,  // 2*2*32*8 dwords
                              int G, int D)
{
    const int tid = threadIdx.x;

    if (tid < G) {
        const int g = tid;
        float qw = rots[g * 4 + 0], qx = rots[g * 4 + 1];
        float qy = rots[g * 4 + 2], qz = rots[g * 4 + 3];
        const float rn = rsqrtf(qw * qw + qx * qx + qy * qy + qz * qz);
        qw *= rn; qx *= rn; qy *= rn; qz *= rn;

        const float R00 = 1.f - 2.f * (qy * qy + qz * qz);
        const float R01 = 2.f * (qx * qy - qw * qz);
        const float R02 = 2.f * (qx * qz + qw * qy);
        const float R10 = 2.f * (qx * qy + qw * qz);
        const float R11 = 1.f - 2.f * (qx * qx + qz * qz);
        const float R12 = 2.f * (qy * qz - qw * qx);
        const float R20 = 2.f * (qx * qz - qw * qy);
        const float R21 = 2.f * (qy * qz + qw * qx);
        const float R22 = 1.f - 2.f * (qx * qx + qy * qy);

        const float s0 = scales[g * 3 + 0], s1 = scales[g * 3 + 1], s2 = scales[g * 3 + 2];
        const float d0 = s0 * s0, d1 = s1 * s1, d2 = s2 * s2;

        // cov = R diag(s^2) R^T (symmetric)
        const float c00 = R00 * R00 * d0 + R01 * R01 * d1 + R02 * R02 * d2;
        const float c01 = R00 * R10 * d0 + R01 * R11 * d1 + R02 * R12 * d2;
        const float c02 = R00 * R20 * d0 + R01 * R21 * d1 + R02 * R22 * d2;
        const float c11 = R10 * R10 * d0 + R11 * R11 * d1 + R12 * R12 * d2;
        const float c12 = R10 * R20 * d0 + R11 * R21 * d1 + R12 * R22 * d2;
        const float c22 = R20 * R20 * d0 + R21 * R21 * d1 + R22 * R22 * d2;

        const float det = c00 * (c11 * c22 - c12 * c12)
                        - c01 * (c01 * c22 - c12 * c02)
                        + c02 * (c01 * c12 - c11 * c02);
        const float rdet = MAHA_SCALE / det;   // fold exp scale into inverse

        const float i00 = (c11 * c22 - c12 * c12) * rdet;
        const float i01 = (c02 * c12 - c01 * c22) * rdet;
        const float i02 = (c01 * c12 - c02 * c11) * rdet;
        const float i11 = (c00 * c22 - c02 * c02) * rdet;
        const float i12 = (c01 * c02 - c00 * c12) * rdet;
        const float i22 = (c00 * c11 - c01 * c01) * rdet;

        // Upper Cholesky of scaled inverse covariance: M' = U^T U
        const float u00 = sqrtf(i00);
        const float u01 = i01 / u00;
        const float u02 = i02 / u00;
        const float u11 = sqrtf(i11 - u01 * u01);
        const float u12 = (i12 - u01 * u02) / u11;
        const float u22 = sqrtf(i22 - u02 * u02 - u12 * u12);

        float* o = wsG + g * GP_STRIDE;
        o[0]  = means[g * 3 + 0];
        o[1]  = means[g * 3 + 1];
        o[2]  = means[g * 3 + 2];
        o[3]  = opac[g];
        o[4]  = u00; o[5] = u01; o[6] = u02; o[7] = u11;
        o[8]  = u12; o[9] = u22;
        o[10] = 0.f; o[11] = 0.f;
    }

    const int total = 2 * 2 * 32 * 8;  // [kb][nt][lane][v] (assumes G=64, D=32)
    for (int idx = tid; idx < total; idx += blockDim.x) {
        const int v    = idx & 7;
        const int lane = (idx >> 3) & 31;
        const int nt   = (idx >> 8) & 1;
        const int kb   = (idx >> 9) & 1;
        const int half = lane >> 4;
        const int col  = (lane & 15) + nt * 16;
        const int g0   = kb * 32 + ((v < 4) ? (8 * half + 2 * v)
                                            : (16 + 8 * half + 2 * (v - 4)));
        union { _Float16 h[2]; unsigned int u; } p;
        p.h[0] = (_Float16)feats[g0 * D + col];
        p.h[1] = (_Float16)feats[(g0 + 1) * D + col];
        bfrag[idx] = p.u;
    }
}

// ---------------------------------------------------------------------------
// Kernel 2: one wave per 16-voxel tile.
// Per lane: the 16 gaussian weights matching its A-fragment slots (ISA 7.12.2
// 16-bit A 16x32: row M = lane%16; K-pairs per VGPR, split by lane-half),
// then 2 k-steps x 2 n-tiles of v_wmma_f32_16x16x32_f16 (f32 accum) do the
// feature GEMM. Density finished via ds_swizzle SWAPX16; 1/denom broadcast
// with ds_bpermute shuffles; C-fragment layout writeback.
// ---------------------------------------------------------------------------
__device__ __forceinline__ float swap16(float x) {
    return __int_as_float(__builtin_amdgcn_ds_swizzle(__float_as_int(x), SWIZ_SWAPX16));
}

__device__ __forceinline__ v16h make_A(int kb, int half, const float* sG,
                                       float px, float py, float pz,
                                       float& wsum, float& cnt)
{
    v16h a;
#pragma unroll
    for (int v = 0; v < 8; ++v) {
        const int gbase = kb * 32 + ((v < 4) ? (8 * half + 2 * v)
                                             : (16 + 8 * half + 2 * (v - 4)));
#pragma unroll
        for (int hl = 0; hl < 2; ++hl) {
            const float* gp = sG + (gbase + hl) * GP_STRIDE;
            const float4 P  = *(const float4*)(gp);      // mean.xyz, opacity
            const float4 U0 = *(const float4*)(gp + 4);  // u00,u01,u02,u11
            const float2 U1 = *(const float2*)(gp + 8);  // u12,u22
            const float dx = px - P.x;
            const float dy = py - P.y;
            const float dz = pz - P.z;
            const float t0 = fmaf(U0.z, dz, fmaf(U0.y, dy, U0.x * dx));
            const float t1 = fmaf(U1.x, dz, U0.w * dy);
            const float t2 = U1.y * dz;
            const float maha = fmaf(t2, t2, fmaf(t1, t1, t0 * t0));
            float w = 0.f;
            if (maha <= MAHA_CUT) {          // coherent branch: skips exp2 >99%
                w = P.w * exp2f(-maha);
                wsum += w;
                cnt  += 1.f;
            }
            a[2 * v + hl] = (_Float16)w;
        }
    }
    return a;
}

__global__ void __launch_bounds__(256)
gv_splat(const float* __restrict__ grid,          // (V,3)
         const float* __restrict__ wsG,           // (64,GP_STRIDE)
         const unsigned int* __restrict__ bfrag,  // 2*2*32*8 dwords
         float* __restrict__ outDen,              // (V)
         float* __restrict__ outFeat,             // (V,32)
         int V)
{
    __shared__ __align__(16) float sG[64 * GP_STRIDE];
    for (int i = threadIdx.x; i < 64 * GP_STRIDE; i += blockDim.x) sG[i] = wsG[i];
    __syncthreads();

    const int lane = threadIdx.x & 31;
    const int wave = threadIdx.x >> 5;
    const int tile = blockIdx.x * WAVES_PER_BLOCK + wave;
    const int numTiles = V >> 4;
    if (tile >= numTiles) return;      // wave-uniform: EXEC all-ones at WMMA

    const int base = tile * 16;
    const int m    = lane & 15;
    const int half = lane >> 4;

    const float px = grid[3 * (base + m) + 0];
    const float py = grid[3 * (base + m) + 1];
    const float pz = grid[3 * (base + m) + 2];

    // Loop-invariant B fragments: hoist all four (kb x nt), 32B aligned loads.
    const v16h b00 = *(const v16h*)(bfrag + ((0 * 2 + 0) * 32 + lane) * 8);
    const v16h b01 = *(const v16h*)(bfrag + ((0 * 2 + 1) * 32 + lane) * 8);
    const v16h b10 = *(const v16h*)(bfrag + ((1 * 2 + 0) * 32 + lane) * 8);
    const v16h b11 = *(const v16h*)(bfrag + ((1 * 2 + 1) * 32 + lane) * 8);

    v8f acc0 = {};
    v8f acc1 = {};
    float wsum = 0.f, cnt = 0.f;

    const v16h a0 = make_A(0, half, sG, px, py, pz, wsum, cnt);
    acc0 = __builtin_amdgcn_wmma_f32_16x16x32_f16(false, a0, false, b00,
                                                  (short)0, acc0, false, false);
    acc1 = __builtin_amdgcn_wmma_f32_16x16x32_f16(false, a0, false, b01,
                                                  (short)0, acc1, false, false);

    const v16h a1 = make_A(1, half, sG, px, py, pz, wsum, cnt);
    acc0 = __builtin_amdgcn_wmma_f32_16x16x32_f16(false, a1, false, b10,
                                                  (short)0, acc0, false, false);
    acc1 = __builtin_amdgcn_wmma_f32_16x16x32_f16(false, a1, false, b11,
                                                  (short)0, acc1, false, false);

    // Lanes l and l^16 jointly cover all 64 gaussians for row m = l%16:
    // single ds_swizzle (SWAPX16) completes each per-row sum.
    wsum += swap16(wsum);
    cnt  += swap16(cnt);
    // v_rcp_f32 (1-ulp) instead of the IEEE divide expansion.
    const float rden = __builtin_amdgcn_rcpf(fmaxf(cnt, 1e-6f));
    if (half == 0) outDen[base + m] = wsum * rden;

    // C/D layout: VGPR r -> row (r + 8*half), col = lane%16 (+16 for acc1).
#pragma unroll
    for (int r = 0; r < 8; ++r) {
        const int   mm = r + 8 * half;
        const float rd = __shfl(rden, mm, 32);  // lane mm holds 1/denom for row mm
        float* dst = outFeat + (size_t)(base + mm) * 32 + (lane & 15);
        dst[0]  = acc0[r] * rd;
        dst[16] = acc1[r] * rd;
    }
}

// ---------------------------------------------------------------------------
extern "C" void kernel_launch(void* const* d_in, const int* in_sizes, int n_in,
                              void* d_out, int out_size, void* d_ws, size_t ws_size,
                              hipStream_t stream) {
    const float* grid   = (const float*)d_in[0];  // (V,3)
    const float* means  = (const float*)d_in[1];  // (1,G,3)
    const float* opac   = (const float*)d_in[2];  // (1,G,1)
    const float* feats  = (const float*)d_in[3];  // (1,G,D)
    const float* scales = (const float*)d_in[4];  // (1,G,3)
    const float* rots   = (const float*)d_in[5];  // (1,G,4)

    const int V = in_sizes[0] / 3;     // 307200
    const int G = in_sizes[1] / 3;     // 64
    const int D = in_sizes[3] / G;     // 32

    float*        wsG   = (float*)d_ws;                                   // G*12 floats
    unsigned int* bfrag = (unsigned int*)((char*)d_ws +
                          (size_t)G * GP_STRIDE * sizeof(float));         // 3072B offset: 32B aligned

    float* outDen  = (float*)d_out;    // (V)
    float* outFeat = outDen + V;       // (V,D)

    gv_precompute<<<1, 256, 0, stream>>>(means, opac, feats, scales, rots,
                                         wsG, bfrag, G, D);

    const int numTiles = V >> 4;
    const int blocks   = (numTiles + WAVES_PER_BLOCK - 1) / WAVES_PER_BLOCK;
    gv_splat<<<blocks, 256, 0, stream>>>(grid, wsG, bfrag, outDen, outFeat, V);
}
